// DeformableMLP_94489281124
// MI455X (gfx1250) — matrix operand order repaired and checked by reading, not compiled, not run on hardware
//
#include <hip/hip_runtime.h>
#include <cmath>

// Problem constants (B=4, C=64, OC=64, H=W=256)
#define Bd  4
#define Cd  64
#define Hd  256
#define Wd  256
#define HWd 65536            // Hd*Wd
#define NTOT (Bd*Cd*HWd)     // 16,777,216 elements
#define NPIX (Bd*HWd)        // 262,144 pixels
#define BN_N 262144.0f

// packed-weight sizes (ushorts): pw = 12 tiles * 2 ksteps * 32 lanes * 16
#define PW_PACK_N 12288
#define W2_PACK_N 4096

typedef __attribute__((ext_vector_type(16))) __bf16 v16bf;
typedef __attribute__((ext_vector_type(8)))  float  v8f;

// ---------- bf16 helpers (storage as ushort to keep host code bf16-free) ----
__device__ __forceinline__ unsigned short f2bf(float f) {
    unsigned u = __float_as_uint(f);
    u += 0x7fffu + ((u >> 16) & 1u);          // round-to-nearest-even
    return (unsigned short)(u >> 16);
}
__device__ __forceinline__ __bf16 us2bf(unsigned short h) {
    union { unsigned short u; __bf16 b; } cv; cv.u = h; return cv.b;
}

// ---------- A-fragment packed load: one contiguous 32B vector per lane ------
__device__ __forceinline__ v16bf load_a_packed(const unsigned short* pk, int mt, int kk, int lane) {
    const unsigned short* p = pk + (((mt * 2 + kk) * 32 + lane) << 4);
    return *(const v16bf*)__builtin_assume_aligned(p, 32);
}

// B: 32x16 (KxN), source bf16 tensor with K-stride = HWd (channel-major image).
// lane n = l&15 picks pixel column; elem e: K = (e<8 ? half*8+e : 16+half*8+(e-8)).
__device__ __forceinline__ v16bf load_b_frag(const unsigned short* base, int c0, int pix, int lane) {
    const int half = lane >> 4;
    v16bf b;
#pragma unroll
    for (int e = 0; e < 16; ++e) {
        const int c = c0 + half * 8 + e + (e >= 8 ? 8 : 0);
        b[e] = us2bf(base[c * HWd + pix]);
    }
    return b;
}

__device__ __forceinline__ float samp(const float* img, int y, int x) {
    return ((unsigned)y < (unsigned)Hd && (unsigned)x < (unsigned)Wd) ? img[(y << 8) + x] : 0.0f;
}

// ==== K0: pack fp32 weights -> bf16, pre-swizzled into WMMA A 16x32 layout ==
// A layout (wave32): lane m = l&15, half = l>>4;
// vector elem j=2v+s : K = (v<4 ? half*8+2v : 16+half*8+2(v-4)) + s
__device__ __forceinline__ void pack_one(const float* __restrict__ W,
                                         unsigned short* __restrict__ dst, int ii) {
    const int j = ii & 15, lane = (ii >> 4) & 31, kk = (ii >> 9) & 1, mt = ii >> 10;
    const int v = j >> 1, s = j & 1, m = lane & 15, half = lane >> 4;
    const int kb = (v < 4) ? (half * 8 + 2 * v + s) : (16 + half * 8 + 2 * (v - 4) + s);
    dst[ii] = f2bf(W[(mt * 16 + m) * 64 + kk * 32 + kb]);
}
__global__ __launch_bounds__(256) void k_pack_weights(const float* __restrict__ pw,
                                                      const float* __restrict__ w2,
                                                      unsigned short* __restrict__ pwpk,
                                                      unsigned short* __restrict__ w2pk) {
    const int i = blockIdx.x * 256 + threadIdx.x;   // 0 .. 16383
    if (i < PW_PACK_N) pack_one(pw, pwpk, i);
    else               pack_one(w2, w2pk, i - PW_PACK_N);
}

// ============ K1: depthwise 3x3, pad=1, f32 -> bf16 workspace ==============
__global__ __launch_bounds__(256) void k_dwconv(const float* __restrict__ x,
                                                const float* __restrict__ wdw,
                                                unsigned short* __restrict__ dwb) {
    const int idx = blockIdx.x * 256 + threadIdx.x;
    const int w = idx & 255, h = (idx >> 8) & 255, c = (idx >> 16) & 63;
    const float* xp = x + (idx & ~0xFFFF);     // base of this (b,c) image plane
    const float* wp = wdw + c * 9;
    if (h < 254) __builtin_prefetch(xp + ((h + 2) << 8) + w, 0, 1);  // global_prefetch_b8
    float acc = 0.f;
#pragma unroll
    for (int dy = -1; dy <= 1; ++dy) {
        const int yy = h + dy;
        if (yy < 0 || yy > 255) continue;
#pragma unroll
        for (int dx = -1; dx <= 1; ++dx) {
            const int xx = w + dx;
            if (xx < 0 || xx > 255) continue;
            acc += xp[(yy << 8) + xx] * wp[(dy + 1) * 3 + (dx + 1)];
        }
    }
    dwb[idx] = f2bf(acc);
}

// ==== K2: WMMA pointwise GEMM (192x64) -> sigmoid/clip -> bilinear sample ===
// One wave per 16-pixel tile. In-place: reads dw, writes sampled*mod (bf16).
__global__ __launch_bounds__(256) void k_offset_sample(const float* __restrict__ x,
                                                       const unsigned short* __restrict__ pwpk,
                                                       unsigned short* __restrict__ dwb) {
    const int gid  = blockIdx.x * blockDim.x + threadIdx.x;
    const int wave = gid >> 5;
    const int lane = threadIdx.x & 31;
    const int w0 = (wave & 15) << 4, h = (wave >> 4) & 255, b = wave >> 12;
    const int n = lane & 15, half = lane >> 4;
    const int pix = (h << 8) + w0 + n;
    unsigned short* base = dwb + b * Cd * HWd;

    const v16bf bf0 = load_b_frag(base, 0,  pix, lane);
    const v16bf bf1 = load_b_frag(base, 32, pix, lane);

#pragma unroll
    for (int t = 0; t < 4; ++t) {
        v8f ay = {}, ax = {}, am = {};
#pragma unroll
        for (int kk = 0; kk < 2; ++kk) {
            const v16bf bb = kk ? bf1 : bf0;
            ay = __builtin_amdgcn_wmma_f32_16x16x32_bf16(false, load_a_packed(pwpk, t,     kk, lane), false, bb, (short)0, ay, false, false);
            ax = __builtin_amdgcn_wmma_f32_16x16x32_bf16(false, load_a_packed(pwpk, 4 + t, kk, lane), false, bb, (short)0, ax, false, false);
            am = __builtin_amdgcn_wmma_f32_16x16x32_bf16(false, load_a_packed(pwpk, 8 + t, kk, lane), false, bb, (short)0, am, false, false);
        }
#pragma unroll
        for (int r = 0; r < 8; ++r) {
            const int c   = t * 16 + r + half * 8;
            const float offy = fminf(fmaxf(ay[r], -64.f), 64.f);
            const float offx = fminf(fmaxf(ax[r], -64.f), 64.f);
            const float md   = 2.0f / (1.0f + __expf(-am[r]));
            const float sy = (float)h + offy, sx = (float)(w0 + n) + offx;
            const float fy = floorf(sy), fx = floorf(sx);
            const float wy = sy - fy, wx = sx - fx;
            const int y0 = (int)fy, x0 = (int)fx;
            const float* img = x + (b * Cd + c) * HWd;
            const float v00 = samp(img, y0,     x0), v01 = samp(img, y0,     x0 + 1);
            const float v10 = samp(img, y0 + 1, x0), v11 = samp(img, y0 + 1, x0 + 1);
            const float val = ((1.f - wy) * ((1.f - wx) * v00 + wx * v01) +
                               wy         * ((1.f - wx) * v10 + wx * v11)) * md;
            base[c * HWd + pix] = f2bf(val);
        }
    }
}

// ============ K3: WMMA 64x64 output GEMM + bias -> f32 d_out ================
__global__ __launch_bounds__(256) void k_outconv(const unsigned short* __restrict__ sampled,
                                                 const unsigned short* __restrict__ w2pk,
                                                 const float* __restrict__ bias,
                                                 float* __restrict__ out) {
    const int gid  = blockIdx.x * blockDim.x + threadIdx.x;
    const int wave = gid >> 5;
    const int lane = threadIdx.x & 31;
    const int w0 = (wave & 15) << 4, h = (wave >> 4) & 255, b = wave >> 12;
    const int n = lane & 15, half = lane >> 4;
    const int pix = (h << 8) + w0 + n;
    const unsigned short* base = sampled + b * Cd * HWd;
    float* obase = out + b * Cd * HWd;

    const v16bf bf0 = load_b_frag(base, 0,  pix, lane);
    const v16bf bf1 = load_b_frag(base, 32, pix, lane);

#pragma unroll
    for (int t = 0; t < 4; ++t) {
        v8f acc = {};
        acc = __builtin_amdgcn_wmma_f32_16x16x32_bf16(false, load_a_packed(w2pk, t, 0, lane), false, bf0, (short)0, acc, false, false);
        acc = __builtin_amdgcn_wmma_f32_16x16x32_bf16(false, load_a_packed(w2pk, t, 1, lane), false, bf1, (short)0, acc, false, false);
#pragma unroll
        for (int r = 0; r < 8; ++r) {
            const int c = t * 16 + r + half * 8;
            obase[c * HWd + pix] = acc[r] + bias[c];
        }
    }
}

// ===== K4: deterministic per-channel BN stats (one block per channel) =======
__global__ __launch_bounds__(256) void k_stats(const float* __restrict__ out,
                                               float* __restrict__ stats) {
    const int c = blockIdx.x, tid = threadIdx.x;
    __shared__ float ss[256], sq[256];
    float s = 0.f, q = 0.f;
    for (int b = 0; b < Bd; ++b) {
        const float* p = out + (b * Cd + c) * HWd;
        for (int i = tid; i < HWd; i += 256) { const float v = p[i]; s += v; q += v * v; }
    }
    ss[tid] = s; sq[tid] = q;
    __syncthreads();
    for (int st = 128; st > 0; st >>= 1) {
        if (tid < st) { ss[tid] += ss[tid + st]; sq[tid] += sq[tid + st]; }
        __syncthreads();
    }
    if (tid == 0) {
        const float mean = ss[0] / BN_N;
        const float var  = sq[0] / BN_N - mean * mean;
        stats[c]      = mean;
        stats[64 + c] = rsqrtf(var + 1e-5f);
    }
}

// ============ K5: BN affine + exact GELU, in-place on d_out =================
__global__ __launch_bounds__(256) void k_bn_gelu(float* __restrict__ out,
                                                 const float* __restrict__ stats,
                                                 const float* __restrict__ gamma,
                                                 const float* __restrict__ beta) {
    const int idx = blockIdx.x * 256 + threadIdx.x;
    const int c = (idx >> 16) & 63;
    const float v = out[idx];
    const float y = (v - stats[c]) * stats[64 + c] * gamma[c] + beta[c];
    out[idx] = 0.5f * y * (1.0f + erff(y * 0.70710678118654752f));
}

extern "C" void kernel_launch(void* const* d_in, const int* in_sizes, int n_in,
                              void* d_out, int out_size, void* d_ws, size_t ws_size,
                              hipStream_t stream) {
    const float* x     = (const float*)d_in[0];   // (4,64,256,256)
    const float* dw_w  = (const float*)d_in[1];   // (64,1,3,3)
    const float* pw_w  = (const float*)d_in[2];   // (192,64)
    const float* w2    = (const float*)d_in[3];   // (64,64)
    const float* bias  = (const float*)d_in[4];   // (64)
    const float* gamma = (const float*)d_in[5];   // (64)
    const float* beta  = (const float*)d_in[6];   // (64)
    float* out = (float*)d_out;

    // workspace layout (all offsets 32B-aligned)
    char* ws = (char*)d_ws;
    unsigned short* dwb   = (unsigned short*)ws;                               // 33,554,432 B (bf16, reused in place)
    float*          stats = (float*)(ws + (size_t)NTOT * 2);                   // 512 B
    unsigned short* pwpk  = (unsigned short*)(ws + (size_t)NTOT * 2 + 512);    // 24,576 B
    unsigned short* w2pk  = pwpk + PW_PACK_N;                                  // 8,192 B

    // 0) pack weights into WMMA A-fragment bf16 layout (once per launch)
    k_pack_weights<<<(PW_PACK_N + W2_PACK_N) / 256, 256, 0, stream>>>(pw_w, w2, pwpk, w2pk);
    // 1) depthwise 3x3 -> bf16
    k_dwconv<<<NTOT / 256, 256, 0, stream>>>(x, dw_w, dwb);
    // 2) pointwise WMMA GEMM + sigmoid/clip + bilinear sample (in place)
    k_offset_sample<<<(NPIX / 16) * 32 / 256, 256, 0, stream>>>(x, pwpk, dwb);
    // 3) output WMMA GEMM + bias
    k_outconv<<<(NPIX / 16) * 32 / 256, 256, 0, stream>>>(dwb, w2pk, bias, out);
    // 4) deterministic BN stats
    k_stats<<<Cd, 256, 0, stream>>>(out, stats);
    // 5) BN + exact GELU
    k_bn_gelu<<<NTOT / 256, 256, 0, stream>>>(out, stats, gamma, beta);
}
